// GAT_66821101191576
// MI455X (gfx1250) — compile-verified, matrix operand order checked by reading, not run on hardware
//
#include <hip/hip_runtime.h>
#include <hip/hip_bf16.h>
#include <stdint.h>

// ---------------- problem constants (from reference) ----------------
#define GN    20000     // nodes
#define GE    640000    // edges (before self-loops)
#define GETOT (GE + GN) // edges incl. self-loops
#define GIN   256       // in feats
#define GH    8         // heads
#define GHF   64        // hidden per head
#define GOUT  32        // out per head
#define NEG_SLOPE 0.2f

typedef __attribute__((ext_vector_type(16))) __bf16 v16bf;
typedef __attribute__((ext_vector_type(8)))  float  v8f;

// ---------------- small helpers ----------------
__global__ void k_fill_f32(float* __restrict__ p, float v, int n) {
    int i = blockIdx.x * blockDim.x + threadIdx.x;
    int stride = gridDim.x * blockDim.x;
    for (; i < n; i += stride) p[i] = v;
}

__global__ void k_cast_bf16(const float* __restrict__ in, __bf16* __restrict__ out, int n) {
    int i = blockIdx.x * blockDim.x + threadIdx.x;
    int stride = gridDim.x * blockDim.x;
    for (; i < n; i += stride) out[i] = (__bf16)in[i];
}

// W [K, Nn] row-major -> Wt [Nn, K] row-major bf16 (so B-fragment K pairs are contiguous)
__global__ void k_transpose_cast(const float* __restrict__ W, __bf16* __restrict__ Wt,
                                 int K, int Nn) {
    int i = blockIdx.x * blockDim.x + threadIdx.x;
    if (i >= K * Nn) return;
    int k = i / Nn, n = i % Nn;
    Wt[(size_t)n * K + k] = (__bf16)W[i];
}

// ---------------- LDS-staged WMMA bf16 GEMM ----------------
// C[M,Nn](bf16) = A[M,K](bf16) * Bt[Nn,K](bf16)^T, f32 accumulate.
// Block: 128 threads (4 waves). Output tile: BM x BN = 32 x 128.
// Wave (wm,wn) computes rows [16*wm,16*wm+16) x cols [64*wn,64*wn+64):
// 4 accumulators per wave -> each A fragment feeds 4 WMMAs.
// B tile (BN x BK bf16) is staged in LDS (padded rows, bank-conflict-free b128 reads).
#define BM  32
#define BN  128
#define BK  32
#define BKP (BK + 8)   // row pad: 80B stride, 16B aligned, conflict-free

__global__ __launch_bounds__(128)
void k_gemm_wmma_bf16(const __bf16* __restrict__ A,   // [M,K] row-major
                      const __bf16* __restrict__ Bt,  // [Nn,K] row-major (transposed B)
                      __bf16* __restrict__ C,         // [M,Nn] row-major, bf16
                      int M, int Nn, int K) {
    __shared__ __bf16 lds_b[BN * BKP];   // 10 KB

    const int tid  = threadIdx.x;
    const int wave = tid >> 5;           // 0..3
    const int lane = tid & 31;           // wave32
    const int wm   = wave & 1;           // M half
    const int wn   = wave >> 1;          // N half (64 cols)
    const int tn   = blockIdx.x;
    const int tm   = blockIdx.y;

    const int mr = lane & 15;            // row in tile (A) / col in tile (B)
    const int kg = (lane >> 4) << 3;     // lane-group K offset: 0 or 8

    const __bf16* __restrict__ arow = A + (size_t)(tm * BM + wm * 16 + mr) * K;

    union F16 { uint4 q[2]; v16bf v; };
    v8f acc[4] = {};

    for (int k0 = 0; k0 < K; k0 += BK) {
        __syncthreads();   // protect LDS from previous iteration's readers
        {   // cooperative stage: thread t copies 64B (col tn*BN+t, K k0..k0+31) to LDS
            const uint4* __restrict__ src =
                (const uint4*)(Bt + (size_t)(tn * BN + tid) * K + k0);
            uint4* dst = (uint4*)(lds_b + tid * BKP);
#pragma unroll
            for (int i = 0; i < 4; ++i) dst[i] = src[i];   // global b128 -> ds b128
        }
        __syncthreads();

        // A fragment: 16-bit A-layout, VGPR v holds K = (v>=4?16:0)+kg+2*(v&3)
        F16 af;
        af.q[0] = *(const uint4*)(arow + k0 + kg);
        af.q[1] = *(const uint4*)(arow + k0 + kg + 16);

#pragma unroll
        for (int j = 0; j < 4; ++j) {
            const int col = wn * 64 + j * 16 + mr;
            F16 bfr;
            bfr.q[0] = *(const uint4*)(lds_b + col * BKP + kg);        // ds_load_b128
            bfr.q[1] = *(const uint4*)(lds_b + col * BKP + kg + 16);
            acc[j] = __builtin_amdgcn_wmma_f32_16x16x32_bf16(
                /*neg_a=*/false, af.v, /*neg_b=*/false, bfr.v,
                /*c_mod=*/(short)0, acc[j], /*reuse_a=*/false, /*reuse_b=*/false);
        }
    }

    // D layout: VGPR r -> M = r + (lane>>4)*8, N = lane&15; store as bf16
    const int rowbase = tm * BM + wm * 16 + ((lane >> 4) << 3);
#pragma unroll
    for (int j = 0; j < 4; ++j) {
        const int col = tn * BN + wn * 64 + j * 16 + (lane & 15);
#pragma unroll
        for (int r = 0; r < 8; ++r)
            C[(size_t)(rowbase + r) * Nn + col] = (__bf16)acc[j][r];
    }
}

// ---------------- per-node attention logits ----------------
// h [n, Hh*C] bf16; asrc/adst [n*Hh] f32
__global__ void k_node_att(const __bf16* __restrict__ h,
                           const float* __restrict__ a_src,
                           const float* __restrict__ a_dst,
                           float* __restrict__ asrc, float* __restrict__ adst,
                           int n, int Hh, int C) {
    int t = blockIdx.x * blockDim.x + threadIdx.x;
    if (t >= n * Hh) return;
    int node = t / Hh, hd = t % Hh;
    const __bf16* __restrict__ row = h + ((size_t)node * Hh + hd) * C;
    const float* __restrict__ as   = a_src + hd * C;
    const float* __restrict__ ad   = a_dst + hd * C;
    float s1 = 0.f, s2 = 0.f;
    for (int c = 0; c < C; c += 8) {
        union { uint4 q; __bf16 e[8]; } u;
        u.q = *(const uint4*)(row + c);
#pragma unroll
        for (int i = 0; i < 8; ++i) {
            float v = (float)u.e[i];
            s1 += v * as[c + i];
            s2 += v * ad[c + i];
        }
    }
    asrc[t] = s1;
    adst[t] = s2;
}

// ---------------- edge helpers ----------------
__device__ __forceinline__ void edge_sd(const int* __restrict__ ei, int e, int& s, int& d) {
    if (e < GE) { s = ei[e]; d = ei[GE + e]; }
    else        { s = e - GE; d = e - GE; }   // self-loops appended
}

__device__ __forceinline__ float lrelu(float v) {
    return v > 0.f ? v : NEG_SLOPE * v;
}

// exact IEEE float atomic-max via signed-max / unsigned-min trick (init = -1e30)
__device__ __forceinline__ void atomicMaxF(float* addr, float val) {
    if (val >= 0.f) atomicMax((int*)addr, __float_as_int(val));
    else            atomicMin((unsigned int*)addr, (unsigned int)__float_as_int(val));
}

__global__ void k_edge_max(const int* __restrict__ ei,
                           const float* __restrict__ asrc, const float* __restrict__ adst,
                           float* __restrict__ mx, int Hh) {
    int t = blockIdx.x * blockDim.x + threadIdx.x;
    if (t >= GETOT * Hh) return;
    int e = t / Hh, hd = t % Hh, s, d;
    edge_sd(ei, e, s, d);
    float v = lrelu(asrc[s * Hh + hd] + adst[d * Hh + hd]);
    atomicMaxF(&mx[d * Hh + hd], v);
}

__global__ void k_edge_exp(const int* __restrict__ ei,
                           const float* __restrict__ asrc, const float* __restrict__ adst,
                           const float* __restrict__ mx, float* __restrict__ den,
                           float* __restrict__ eattn, int Hh) {
    int t = blockIdx.x * blockDim.x + threadIdx.x;
    if (t >= GETOT * Hh) return;
    int e = t / Hh, hd = t % Hh, s, d;
    edge_sd(ei, e, s, d);
    float v  = lrelu(asrc[s * Hh + hd] + adst[d * Hh + hd]);
    float ex = __expf(v - mx[d * Hh + hd]);
    eattn[t] = ex;
    atomicAdd(&den[d * Hh + hd], ex);
}

// weighted scatter-aggregate: agg[dst,h,:] += alpha * h[src,h,:]  (h is bf16, agg f32)
__global__ void k_edge_agg(const int* __restrict__ ei,
                           const float* __restrict__ eattn, const float* __restrict__ den,
                           const __bf16* __restrict__ hfeat, float* __restrict__ agg,
                           int Hh, int C) {
    int t = blockIdx.x * blockDim.x + threadIdx.x;
    if (t >= GETOT * Hh) return;
    int e = t / Hh, hd = t % Hh, s, d;
    edge_sd(ei, e, s, d);
    float alpha = eattn[t] / den[d * Hh + hd];
    const __bf16* __restrict__ hrow = hfeat + ((size_t)s * Hh + hd) * C;
    float*        __restrict__ arow = agg   + ((size_t)d * Hh + hd) * C;
    __builtin_prefetch(hrow, 0, 1);   // global_prefetch_b8 on the gathered row
    for (int c = 0; c < C; c += 8) {
        union { uint4 q; __bf16 e8[8]; } u;
        u.q = *(const uint4*)(hrow + c);
#pragma unroll
        for (int i = 0; i < 8; ++i)
            atomicAdd(arow + c + i, alpha * (float)u.e8[i]);
    }
}

// mean over heads + bias + relu -> bf16 (layer-1 epilogue / layer-2 GEMM input)
__global__ void k_finalize1(const float* __restrict__ agg, const float* __restrict__ b,
                            __bf16* __restrict__ out, int n, int Hh, int C) {
    int t = blockIdx.x * blockDim.x + threadIdx.x;
    if (t >= n * C) return;
    int node = t / C, c = t % C;
    float s = 0.f;
    for (int h = 0; h < Hh; ++h) s += agg[((size_t)node * Hh + h) * C + c];
    s = s / (float)Hh + b[c];
    s = s > 0.f ? s : 0.f;
    out[t] = (__bf16)s;
}

// mean over heads + bias -> f32 (final output)
__global__ void k_finalize2(const float* __restrict__ agg, const float* __restrict__ b,
                            float* __restrict__ out, int n, int Hh, int C) {
    int t = blockIdx.x * blockDim.x + threadIdx.x;
    if (t >= n * C) return;
    int node = t / C, c = t % C;
    float s = 0.f;
    for (int h = 0; h < Hh; ++h) s += agg[((size_t)node * Hh + h) * C + c];
    out[t] = s / (float)Hh + b[c];
}

// ---------------- launch ----------------
extern "C" void kernel_launch(void* const* d_in, const int* in_sizes, int n_in,
                              void* d_out, int out_size, void* d_ws, size_t ws_size,
                              hipStream_t stream) {
    const float* x        = (const float*)d_in[0];
    const int*   ei       = (const int*)  d_in[1];
    const float* W1       = (const float*)d_in[2];
    const float* att_src1 = (const float*)d_in[3];
    const float* att_dst1 = (const float*)d_in[4];
    const float* b1       = (const float*)d_in[5];
    const float* W2       = (const float*)d_in[6];
    const float* att_src2 = (const float*)d_in[7];
    const float* att_dst2 = (const float*)d_in[8];
    const float* b2       = (const float*)d_in[9];
    float* out = (float*)d_out;

    // ---- workspace carve-up (256B aligned), total ~99 MB ----
    char* base = (char*)d_ws;
    size_t off = 0;
    auto carve = [&](size_t bytes) -> char* {
        char* p = base + off;
        off += (bytes + 255) & ~(size_t)255;
        return p;
    };
    __bf16* xb    = (__bf16*)carve((size_t)GN * GIN * 2);            // bf16 input
    __bf16* w1t   = (__bf16*)carve((size_t)(GH * GHF) * GIN * 2);    // W1^T bf16 [512,256]
    __bf16* h1    = (__bf16*)carve((size_t)GN * GH * GHF * 2);       // 20.5 MB; reused as h3
    float*  asrc  = (float*) carve((size_t)GN * GH * 4);
    float*  adst  = (float*) carve((size_t)GN * GH * 4);
    float*  mx    = (float*) carve((size_t)GN * GH * 4);
    float*  den   = (float*) carve((size_t)GN * GH * 4);
    float*  eattn = (float*) carve((size_t)GETOT * GH * 4);          // 21 MB
    float*  agg   = (float*) carve((size_t)GN * GH * GHF * 4);       // 41 MB; reused layer 2
    __bf16* h2b   = (__bf16*)carve((size_t)GN * GHF * 2);            // layer-1 output bf16
    __bf16* w2t   = (__bf16*)carve((size_t)(GH * GOUT) * GHF * 2);   // W2^T bf16 [256,64]
    __bf16* h3    = h1;                                              // [20000,256] bf16 fits

    const int TB = 256;
    const int edgeBlocks = (GETOT * GH + TB - 1) / TB;

    // ===================== Layer 1 =====================
    k_cast_bf16<<<2048, TB, 0, stream>>>(x, xb, GN * GIN);
    k_transpose_cast<<<(GIN * GH * GHF + TB - 1) / TB, TB, 0, stream>>>(W1, w1t, GIN, GH * GHF);

    {   // h1[20000,512] = xb @ W1 (bf16 out, f32 accumulate)
        dim3 grid((GH * GHF) / BN, GN / BM);
        k_gemm_wmma_bf16<<<grid, 128, 0, stream>>>(xb, w1t, h1, GN, GH * GHF, GIN);
    }

    k_node_att<<<(GN * GH + TB - 1) / TB, TB, 0, stream>>>(h1, att_src1, att_dst1,
                                                           asrc, adst, GN, GH, GHF);
    k_fill_f32<<<256, TB, 0, stream>>>(mx,  -1e30f, GN * GH);
    k_fill_f32<<<256, TB, 0, stream>>>(den,  0.0f,  GN * GH);
    k_fill_f32<<<4096, TB, 0, stream>>>(agg, 0.0f,  GN * GH * GHF);

    k_edge_max<<<edgeBlocks, TB, 0, stream>>>(ei, asrc, adst, mx, GH);
    k_edge_exp<<<edgeBlocks, TB, 0, stream>>>(ei, asrc, adst, mx, den, eattn, GH);
    k_edge_agg<<<edgeBlocks, TB, 0, stream>>>(ei, eattn, den, h1, agg, GH, GHF);

    k_finalize1<<<(GN * GHF + TB - 1) / TB, TB, 0, stream>>>(agg, b1, h2b, GN, GH, GHF);

    // ===================== Layer 2 =====================
    k_transpose_cast<<<(GHF * GH * GOUT + TB - 1) / TB, TB, 0, stream>>>(W2, w2t, GHF, GH * GOUT);

    {   // h3[20000,256] = h2b @ W2   (h3 aliases h1's slot; h1 dead now)
        dim3 grid((GH * GOUT) / BN, GN / BM);
        k_gemm_wmma_bf16<<<grid, 128, 0, stream>>>(h2b, w2t, h3, GN, GH * GOUT, GHF);
    }

    k_node_att<<<(GN * GH + TB - 1) / TB, TB, 0, stream>>>(h3, att_src2, att_dst2,
                                                           asrc, adst, GN, GH, GOUT);
    k_fill_f32<<<256, TB, 0, stream>>>(mx,  -1e30f, GN * GH);
    k_fill_f32<<<256, TB, 0, stream>>>(den,  0.0f,  GN * GH);
    k_fill_f32<<<4096, TB, 0, stream>>>(agg, 0.0f,  GN * GH * GOUT);

    k_edge_max<<<edgeBlocks, TB, 0, stream>>>(ei, asrc, adst, mx, GH);
    k_edge_exp<<<edgeBlocks, TB, 0, stream>>>(ei, asrc, adst, mx, den, eattn, GH);
    k_edge_agg<<<edgeBlocks, TB, 0, stream>>>(ei, eattn, den, h3, agg, GH, GOUT);

    k_finalize2<<<(GN * GOUT + TB - 1) / TB, TB, 0, stream>>>(agg, b2, out, GN, GH, GOUT);
}
